// VectorQuantizer_17841294148021
// MI455X (gfx1250) — matrix operand order repaired and checked by reading, not compiled, not run on hardware
//
#include <hip/hip_runtime.h>
#include <hip/hip_bf16.h>
#include <math.h>

// Problem constants (from reference)
#define NN 16384
#define KK 2048
#define DD 1000
#define DP 1024                       // padded K-dim (zero tail)
#define DCH 32                        // 32 d-chunks of 32
#define NSPLIT 4
#define COLS_PER_SPLIT (KK / NSPLIT)  // 512
#define NBLK (COLS_PER_SPLIT / 64)    // 8 N-blocks of 64 per split
#define PITCH 40                      // LDS row pitch in bf16 elems (80B = 20 banks)
#define PSEL 0x07060302u              // v_perm sel: {hi16(src0), hi16(src1)}

typedef __attribute__((ext_vector_type(16))) __bf16 v16bf;
typedef __attribute__((ext_vector_type(8)))  float  v8f;

union FragBF { uint4 q[2]; v16bf v; };

// ---------------------------------------------------------------------------
// Kernel: zero histogram + loss accumulators in workspace
// ---------------------------------------------------------------------------
__global__ void vq_zero_small(int* __restrict__ hist, double* __restrict__ sums) {
  for (int i = threadIdx.x; i < KK; i += 256) hist[i] = 0;
  if (threadIdx.x == 0) { sums[0] = 0.0; sums[1] = 0.0; }
}

// ---------------------------------------------------------------------------
// Kernel: pre-convert W -> bf16 hi/lo (truncation split), padded to DP.
// One block per code row; 256 threads x 4 elems.
// ---------------------------------------------------------------------------
__global__ __launch_bounds__(256) void vq_wconvert(const float* __restrict__ W,
                                                   unsigned short* __restrict__ Wh,
                                                   unsigned short* __restrict__ Wl) {
  const int k = blockIdx.x;
  const int d0 = threadIdx.x * 4;
  float v[4];
  if (d0 + 4 <= DD) {
    float4 f = *(const float4*)(W + (size_t)k * DD + d0);
    v[0] = f.x; v[1] = f.y; v[2] = f.z; v[3] = f.w;
  } else {
#pragma unroll
    for (int j = 0; j < 4; ++j) v[j] = (d0 + j < DD) ? W[(size_t)k * DD + d0 + j] : 0.0f;
  }
  unsigned u[4], r[4];
#pragma unroll
  for (int j = 0; j < 4; ++j) {
    u[j] = __float_as_uint(v[j]);
    r[j] = __float_as_uint(v[j] - __uint_as_float(u[j] & 0xffff0000u));
  }
  uint2 hq, lq;
  hq.x = __builtin_amdgcn_perm(u[1], u[0], PSEL);
  hq.y = __builtin_amdgcn_perm(u[3], u[2], PSEL);
  lq.x = __builtin_amdgcn_perm(r[1], r[0], PSEL);
  lq.y = __builtin_amdgcn_perm(r[3], r[2], PSEL);
  *(uint2*)(Wh + (size_t)k * DP + d0) = hq;
  *(uint2*)(Wl + (size_t)k * DP + d0) = lq;
}

// ---------------------------------------------------------------------------
// Kernel: exact f32 |w_k|^2
// ---------------------------------------------------------------------------
__global__ __launch_bounds__(128) void vq_wnorm(const float* __restrict__ W,
                                                float* __restrict__ wnorm2) {
  __shared__ float s[128];
  const int k = blockIdx.x;
  const float* w = W + (size_t)k * DD;
  float a = 0.f;
  for (int d = threadIdx.x; d < DD; d += 128) { float v = w[d]; a += v * v; }
  s[threadIdx.x] = a;
  __syncthreads();
  for (int off = 64; off > 0; off >>= 1) {
    if (threadIdx.x < off) s[threadIdx.x] += s[threadIdx.x + off];
    __syncthreads();
  }
  if (threadIdx.x == 0) wnorm2[k] = s[0];
}

// ---------------------------------------------------------------------------
// Main kernel: split-bf16 WMMA GEMM (X @ W^T) fused with running argmin of
// d_k = |w_k|^2 - 2*x.w_k.  Grid: (NN/128, NSPLIT), 256 threads (8 waves).
// A (X) converted on the fly into double-buffered LDS; B fragments come
// straight from the pre-converted, L2-resident Wh/Wl.
// ---------------------------------------------------------------------------
#define AH_(b) ((unsigned short*)(smem +     0 + (b) * 10240))
#define AL_(b) ((unsigned short*)(smem + 20480 + (b) * 10240))

__global__ __launch_bounds__(256) void vq_gemm_argmin(
    const float* __restrict__ X,
    const unsigned short* __restrict__ Wh, const unsigned short* __restrict__ Wl,
    const float* __restrict__ wnorm2,
    float* __restrict__ pval, int* __restrict__ pidx) {
  __shared__ __align__(16) unsigned char smem[40960];  // A hi/lo, 2 buffers

  const int tid  = threadIdx.x;
  const int lane = tid & 31;
  const int wv   = tid >> 5;     // wave 0..7
  const int mi   = wv >> 1;      // 0..3 : 32-row band
  const int ni   = wv & 1;       // 0..1 : 32-col band
  const int l15  = lane & 15;
  const int kh   = lane >> 4;    // K-half per WMMA register layout

  const int rowbase = blockIdx.x * 128;
  const int split   = blockIdx.y;

  // staging roles (A only)
  const int arow  = tid >> 1;    // 0..127, 16 d-values each
  const int ahalf = tid & 1;

  float minv[2][8];
  int   mini[2][8];
#pragma unroll
  for (int ms = 0; ms < 2; ++ms)
#pragma unroll
    for (int r = 0; r < 8; ++r) { minv[ms][r] = 3.4e38f; mini[ms][r] = 0; }

  float ar[16];

  for (int nb = 0; nb < NBLK; ++nb) {
    const int colbase = split * COLS_PER_SPLIT + nb * 64;

    v8f acc[2][2] = {};

    auto stage_load = [&](int dc) {
      const int d0 = dc * 32;
      const float* gx = X + (size_t)(rowbase + arow) * DD + d0 + ahalf * 16;
      if (dc < 31) {
        float4 a0 = *(const float4*)(gx + 0);
        float4 a1 = *(const float4*)(gx + 4);
        float4 a2 = *(const float4*)(gx + 8);
        float4 a3 = *(const float4*)(gx + 12);
        ar[0]  = a0.x; ar[1]  = a0.y; ar[2]  = a0.z; ar[3]  = a0.w;
        ar[4]  = a1.x; ar[5]  = a1.y; ar[6]  = a1.z; ar[7]  = a1.w;
        ar[8]  = a2.x; ar[9]  = a2.y; ar[10] = a2.z; ar[11] = a2.w;
        ar[12] = a3.x; ar[13] = a3.y; ar[14] = a3.z; ar[15] = a3.w;
      } else {
#pragma unroll
        for (int j = 0; j < 16; ++j) {
          int d = d0 + ahalf * 16 + j;
          ar[j] = (d < DD) ? gx[j] : 0.0f;
        }
      }
    };

    auto stage_store = [&](int b) {
      // truncation split: hi = top16(x); lo = top16(x - hi)  (~2^-17 rel err)
      unsigned hi[8], lo[8];
#pragma unroll
      for (int j = 0; j < 8; ++j) {
        unsigned u0 = __float_as_uint(ar[2 * j]);
        unsigned u1 = __float_as_uint(ar[2 * j + 1]);
        unsigned r0 = __float_as_uint(ar[2 * j]     - __uint_as_float(u0 & 0xffff0000u));
        unsigned r1 = __float_as_uint(ar[2 * j + 1] - __uint_as_float(u1 & 0xffff0000u));
        hi[j] = __builtin_amdgcn_perm(u1, u0, PSEL);
        lo[j] = __builtin_amdgcn_perm(r1, r0, PSEL);
      }
      uint4* pah = (uint4*)(AH_(b) + arow * PITCH + ahalf * 16);
      pah[0] = make_uint4(hi[0], hi[1], hi[2], hi[3]);
      pah[1] = make_uint4(hi[4], hi[5], hi[6], hi[7]);
      uint4* pal = (uint4*)(AL_(b) + arow * PITCH + ahalf * 16);
      pal[0] = make_uint4(lo[0], lo[1], lo[2], lo[3]);
      pal[1] = make_uint4(lo[4], lo[5], lo[6], lo[7]);
    };

    // prologue: stage chunk 0 into buffer 0
    stage_load(0);
    stage_store(0);
    __syncthreads();
    int buf = 0;

    for (int dc = 0; dc < DCH; ++dc) {
      const bool more = (dc + 1) < DCH;
      const int d0 = dc * 32;
      if (more) stage_load(dc + 1);   // global loads overlap WMMA below

      // B fragments straight from L2-resident pre-converted W
      FragBF fBh[2], fBl[2];
#pragma unroll
      for (int nf = 0; nf < 2; ++nf) {
        const int code = colbase + ni * 32 + nf * 16 + l15;
        const unsigned short* pb = Wh + (size_t)code * DP + d0 + kh * 16;
        fBh[nf].q[0] = *(const uint4*)(pb);
        fBh[nf].q[1] = *(const uint4*)(pb + 8);
        const unsigned short* pc = Wl + (size_t)code * DP + d0 + kh * 16;
        fBl[nf].q[0] = *(const uint4*)(pc);
        fBl[nf].q[1] = *(const uint4*)(pc + 8);
      }
      // A fragments from LDS
      FragBF fAh[2], fAl[2];
#pragma unroll
      for (int ms = 0; ms < 2; ++ms) {
        const int rowA = mi * 32 + ms * 16 + l15;
        const unsigned short* pa = AH_(buf) + rowA * PITCH;
        fAh[ms].q[0] = *(const uint4*)(pa + kh * 16);
        fAh[ms].q[1] = *(const uint4*)(pa + kh * 16 + 8);
        const unsigned short* pl = AL_(buf) + rowA * PITCH;
        fAl[ms].q[0] = *(const uint4*)(pl + kh * 16);
        fAl[ms].q[1] = *(const uint4*)(pl + kh * 16 + 8);
      }

      // split-bf16 product: Ah*Bh + Al*Bh + Ah*Bl  (~fp32 accuracy)
#pragma unroll
      for (int ms = 0; ms < 2; ++ms)
#pragma unroll
        for (int nf = 0; nf < 2; ++nf) {
          acc[ms][nf] = __builtin_amdgcn_wmma_f32_16x16x32_bf16(
              false, fAh[ms].v, false, fBh[nf].v, (short)0, acc[ms][nf], false, false);
          acc[ms][nf] = __builtin_amdgcn_wmma_f32_16x16x32_bf16(
              false, fAl[ms].v, false, fBh[nf].v, (short)0, acc[ms][nf], false, false);
          acc[ms][nf] = __builtin_amdgcn_wmma_f32_16x16x32_bf16(
              false, fAh[ms].v, false, fBl[nf].v, (short)0, acc[ms][nf], false, false);
        }

      if (more) stage_store(buf ^ 1);
      __syncthreads();
      buf ^= 1;
    }

    // distance tile + running argmin update
#pragma unroll
    for (int nf = 0; nf < 2; ++nf) {
      const int code = colbase + ni * 32 + nf * 16 + l15;
      const float w2 = wnorm2[code];
#pragma unroll
      for (int ms = 0; ms < 2; ++ms)
#pragma unroll
        for (int r = 0; r < 8; ++r) {
          float dist = w2 - 2.0f * acc[ms][nf][r];
          if (dist < minv[ms][r]) { minv[ms][r] = dist; mini[ms][r] = code; }
        }
    }
  }

  // cross-lane argmin reduction (reuse staging LDS as scratch: 32KB <= 40KB)
  __syncthreads();
  float* rv = (float*)smem;             // [128 rows][32 slots]
  int*   ri = (int*)(smem + 16384);
#pragma unroll
  for (int ms = 0; ms < 2; ++ms)
#pragma unroll
    for (int r = 0; r < 8; ++r) {
      const int rho = mi * 32 + ms * 16 + kh * 8 + r;  // row within WG (0..127)
      rv[rho * 32 + ni * 16 + l15] = minv[ms][r];
      ri[rho * 32 + ni * 16 + l15] = mini[ms][r];
    }
  __syncthreads();
  if (tid < 128) {
    float bv = rv[tid * 32];
    int   bi = ri[tid * 32];
#pragma unroll 4
    for (int c = 1; c < 32; ++c) {
      float v = rv[tid * 32 + c];
      int   ii = ri[tid * 32 + c];
      if (v < bv) { bv = v; bi = ii; }
    }
    pval[(size_t)split * NN + rowbase + tid] = bv;
    pidx[(size_t)split * NN + rowbase + tid] = bi;
  }
}

// ---------------------------------------------------------------------------
// Kernel: reduce NSPLIT argmin candidates per row -> close_idx
// ---------------------------------------------------------------------------
__global__ void vq_reduce(const float* __restrict__ pval, const int* __restrict__ pidx,
                          int* __restrict__ close) {
  const int n = blockIdx.x * blockDim.x + threadIdx.x;
  if (n >= NN) return;
  float bv = pval[n];
  int   bi = pidx[n];
#pragma unroll
  for (int s = 1; s < NSPLIT; ++s) {
    float v = pval[(size_t)s * NN + n];
    int   i = pidx[(size_t)s * NN + n];
    if (v < bv) { bv = v; bi = i; }
  }
  close[n] = bi;
}

// ---------------------------------------------------------------------------
// Kernel: per-row gather + quantized_st output + loss partial sums + histogram
// ---------------------------------------------------------------------------
__global__ __launch_bounds__(128) void vq_finalize(
    const float* __restrict__ X, const float* __restrict__ W,
    const int* __restrict__ label, const int* __restrict__ close,
    float* __restrict__ outq, double* __restrict__ sums, int* __restrict__ hist) {
  __shared__ float s1[128];
  __shared__ float s2[128];
  const int n = blockIdx.x;
  const int lbl = label[n];
  const int cls = close[n];
  const float* x  = X + (size_t)n * DD;
  const float* eL = W + (size_t)lbl * DD;
  const float* eC = W + (size_t)cls * DD;
  float q = 0.f, xl = 0.f;
  for (int d = threadIdx.x; d < DD; d += 128) {
    float xv = x[d];
    float ql = eL[d];
    outq[(size_t)n * DD + d] = ql;      // straight-through forward value
    float dq = ql - xv;
    q += dq * dq;
    float dc = eC[d] - xv;
    xl += dc * dc;
  }
  s1[threadIdx.x] = q;
  s2[threadIdx.x] = xl;
  __syncthreads();
  for (int off = 64; off > 0; off >>= 1) {
    if (threadIdx.x < off) {
      s1[threadIdx.x] += s1[threadIdx.x + off];
      s2[threadIdx.x] += s2[threadIdx.x + off];
    }
    __syncthreads();
  }
  if (threadIdx.x == 0) {
    atomicAdd(&sums[0], (double)s1[0]);
    if (lbl != cls) atomicAdd(&sums[1], (double)s2[0]);
    atomicAdd(&hist[lbl], 1);
  }
}

// ---------------------------------------------------------------------------
// Kernels: one-hot encodings (zero then scatter) — enc base is only 8B aligned
// ---------------------------------------------------------------------------
__global__ void vq_zero_enc(float2* __restrict__ enc2) {
  const size_t total = (size_t)NN * KK / 2;
  const size_t stride = (size_t)gridDim.x * blockDim.x;
  for (size_t i = (size_t)blockIdx.x * blockDim.x + threadIdx.x; i < total; i += stride)
    enc2[i] = make_float2(0.f, 0.f);
}
__global__ void vq_scatter(const int* __restrict__ label, float* __restrict__ enc) {
  const int n = blockIdx.x * blockDim.x + threadIdx.x;
  if (n < NN) enc[(size_t)n * KK + label[n]] = 1.0f;
}

// ---------------------------------------------------------------------------
// Kernel: scalars — loss and perplexity
// ---------------------------------------------------------------------------
__global__ void vq_scalars(const int* __restrict__ hist, const double* __restrict__ sums,
                           float* __restrict__ out_loss, float* __restrict__ out_perp) {
  __shared__ float s[256];
  float e = 0.f;
  for (int k = threadIdx.x; k < KK; k += 256) {
    float p = (float)hist[k] * (1.0f / (float)NN);
    e += p * logf(p + 1e-10f);
  }
  s[threadIdx.x] = e;
  __syncthreads();
  for (int off = 128; off > 0; off >>= 1) {
    if (threadIdx.x < off) s[threadIdx.x] += s[threadIdx.x + off];
    __syncthreads();
  }
  if (threadIdx.x == 0) {
    *out_perp = expf(-s[0]);
    const double inv = 1.0 / ((double)NN * (double)DD);
    // (q + 0.25 e) - (x + 0.1 d) with forward-equal pairs
    *out_loss = (float)(1.25 * sums[0] * inv - 1.10 * sums[1] * inv);
  }
}

// ---------------------------------------------------------------------------
extern "C" void kernel_launch(void* const* d_in, const int* in_sizes, int n_in,
                              void* d_out, int out_size, void* d_ws, size_t ws_size,
                              hipStream_t stream) {
  (void)in_sizes; (void)n_in; (void)out_size; (void)ws_size;
  const float* X   = (const float*)d_in[0];   // [N, D] f32
  const int* label = (const int*)d_in[1];     // [N] i32
  const float* W   = (const float*)d_in[2];   // [K, D] f32
  float* out = (float*)d_out;

  // output layout: [loss][quantized_st N*D][perplexity][encodings N*K]
  float* out_loss = out;
  float* out_q    = out + 1;
  float* out_perp = out + 1 + (size_t)NN * DD;
  float* out_enc  = out + 2 + (size_t)NN * DD;

  // workspace carve (~9.0 MB)
  char* ws = (char*)d_ws;
  unsigned short* Wh = (unsigned short*)(ws + 0);         // 4 MB
  unsigned short* Wl = (unsigned short*)(ws + 4194304);   // 4 MB
  float*  wnorm2 = (float*)(ws + 8388608);                // 8 KB
  float*  pval   = (float*)(ws + 8396800);                // 256 KB
  int*    pidx   = (int*)(ws + 8658944);                  // 256 KB
  int*    close  = (int*)(ws + 8921088);                  // 64 KB
  int*    hist   = (int*)(ws + 8986624);                  // 8 KB
  double* sums   = (double*)(ws + 8994816);               // 16 B

  vq_zero_small<<<1, 256, 0, stream>>>(hist, sums);
  vq_wconvert<<<KK, 256, 0, stream>>>(W, Wh, Wl);
  vq_wnorm<<<KK, 128, 0, stream>>>(W, wnorm2);
  vq_gemm_argmin<<<dim3(NN / 128, NSPLIT), 256, 0, stream>>>(X, Wh, Wl, wnorm2, pval, pidx);
  vq_reduce<<<NN / 256, 256, 0, stream>>>(pval, pidx, close);
  vq_finalize<<<NN, 128, 0, stream>>>(X, W, label, close, out_q, sums, hist);
  vq_zero_enc<<<8192, 256, 0, stream>>>((float2*)out_enc);
  vq_scatter<<<NN / 256, 256, 0, stream>>>(label, out_enc);
  vq_scalars<<<1, 256, 0, stream>>>(hist, sums, out_loss, out_perp);
}
